// LinearAttention_9431748182266
// MI455X (gfx1250) — matrix operand order repaired and checked by reading, not compiled, not run on hardware
//
#include <hip/hip_runtime.h>
#include <cstdint>

typedef __bf16 bf16;
typedef __attribute__((ext_vector_type(16))) __bf16 v16bf;
typedef __attribute__((ext_vector_type(8)))  __bf16 v8bf;
typedef __attribute__((ext_vector_type(8)))  float  v8f;
typedef __attribute__((ext_vector_type(4)))  unsigned int u32x4;
typedef __attribute__((ext_vector_type(8)))  int i32x8;
typedef __attribute__((ext_vector_type(4)))  int i32x4;

#define B_ 8
#define S_ 4096
#define D_ 1024
#define F_ 1024
#define EPS_ 1e-6f

#define BM 128
#define BN 64
#define BK 32
#define NTHREADS 256
#define LDSTR 40   // padded LDS row stride in bf16 (80 bytes, 16B aligned)

// ---- WMMA helpers -----------------------------------------------------------

__device__ __forceinline__ v8f wmma_bf16f32(v16bf a, v16bf b, v8f c) {
  return __builtin_amdgcn_wmma_f32_16x16x32_bf16(
      /*neg_a=*/false, a, /*neg_b=*/false, b,
      /*c_mod=*/(short)0, c, /*reuse_a=*/false, /*reuse_b=*/false);
}

// A fragment (16x32 bf16): lane holds M=lane&15, K = {h*8..+7, 16+h*8..+7}
__device__ __forceinline__ v16bf frag_a(const bf16* base) {
  const int lane = threadIdx.x & 31;
  const bf16* p = base + (lane & 15) * LDSTR + (lane >> 4) * 8;
  v8bf lo = *(const v8bf*)(p);
  v8bf hi = *(const v8bf*)(p + 16);
  return __builtin_shufflevector(lo, hi, 0,1,2,3,4,5,6,7,8,9,10,11,12,13,14,15);
}

// B fragment (32x16 bf16, LDS stored N-major): lane holds N=lane&15,
// K = {h*8..+7, 16+h*8..+7}
__device__ __forceinline__ v16bf frag_b(const bf16* base) {
  const int lane = threadIdx.x & 31;
  const bf16* p = base + (lane & 15) * LDSTR + (lane >> 4) * 8;
  v8bf lo = *(const v8bf*)(p);
  v8bf hi = *(const v8bf*)(p + 16);
  return __builtin_shufflevector(lo, hi, 0,1,2,3,4,5,6,7,8,9,10,11,12,13,14,15);
}

// ---- Tensor Data Mover: 2-D bf16 tile -> LDS --------------------------------
// Rows of tile_d0 (=32) bf16 = 64B; pad_interval=3 (64B) + pad_amount=3 (16B)
// lands each row at an 80B stride == LDSTR.
__device__ __forceinline__ void tdm_load_tile_2d(uint32_t lds_byte_off,
                                                 const bf16* gtile,
                                                 uint32_t tensor_d0,
                                                 uint32_t tensor_d1,
                                                 uint32_t tile_d0,
                                                 uint32_t tile_d1,
                                                 uint32_t d0_stride) {
  const uint64_t ga = (uint64_t)(uintptr_t)gtile;
  u32x4 g0;
  g0.x = 1u;                                   // count=1, user descriptor
  g0.y = lds_byte_off;                         // lds_addr
  g0.z = (uint32_t)(ga & 0xFFFFFFFFu);         // global_addr[31:0]
  g0.w = (uint32_t)((ga >> 32) & 0x01FFFFFFu)  // global_addr[56:32]
         | (2u << 30);                         // type=2 ("image")
  i32x8 g1;
  g1[0] = (int)((1u << 16)                     // data_size = 1 -> 2 bytes
                | (1u << 20)                   // pad_enable
                | (3u << 22)                   // pad_interval: 64B
                | (3u << 25));                 // pad_amount: 4 dwords = 16B
  g1[1] = (int)((tensor_d0 & 0xFFFFu) << 16);  // tensor_dim0 lo16
  g1[2] = (int)(((tensor_d0 >> 16) & 0xFFFFu) |
                ((tensor_d1 & 0xFFFFu) << 16));
  g1[3] = (int)(((tensor_d1 >> 16) & 0xFFFFu) |
                ((tile_d0 & 0xFFFFu) << 16));  // tile_dim0
  g1[4] = (int)(tile_d1 & 0xFFFFu);            // tile_dim1 (tile_dim2 = 0)
  g1[5] = (int)d0_stride;                      // tensor_dim0_stride lo32
  g1[6] = 0;
  g1[7] = 0;
  const i32x4 z4 = {0, 0, 0, 0};
  const i32x8 z8 = {0, 0, 0, 0, 0, 0, 0, 0};
  // 6-arg form (clang-23 / therock-10.0 toolchain)
  __builtin_amdgcn_tensor_load_to_lds(g0, g1, z4, z4, z8, 0);
}

__device__ __forceinline__ uint32_t lds_off(const void* p) {
  return (uint32_t)(uintptr_t)p;   // flat shared addr: low 32 bits = LDS offset
}

// ---- Kernel 1: Q/K/V projection, fused relu(.)+1 for Q and K ----------------
__global__ __launch_bounds__(NTHREADS)
void qkv_proj_kernel(const float* __restrict__ x,
                     const float* __restrict__ Wq,
                     const float* __restrict__ Wk,
                     const float* __restrict__ Wv,
                     bf16* __restrict__ Qb, bf16* __restrict__ Kb,
                     bf16* __restrict__ Vb)
{
  __shared__ bf16 sA[BM * LDSTR];
  __shared__ bf16 sB[BN * LDSTR];

  const int t = threadIdx.x;
  const int z = blockIdx.z;
  const float* W = (z == 0) ? Wq : (z == 1) ? Wk : Wv;
  bf16* out      = (z == 0) ? Qb : (z == 1) ? Kb : Vb;
  const bool act = (z < 2);

  const int row0 = blockIdx.y * BM;   // over B*S
  const int col0 = blockIdx.x * BN;   // over F

  const int w = t >> 5, wm = w & 3, wn = w >> 2, lane = t & 31;
  v8f acc[2][2] = {};

  for (int kt = 0; kt < D_ / BK; ++kt) {
    if (kt) __syncthreads();
    // A tile: x[row0..+128][kt*32..+32], fp32 -> bf16 (b128 loads)
    {
      const int ar = t >> 1, ac = (t & 1) * 16;
      const float4* src =
          (const float4*)(x + (size_t)(row0 + ar) * D_ + kt * BK + ac);
      bf16* dst = sA + ar * LDSTR + ac;
#pragma unroll
      for (int j = 0; j < 4; ++j) {
        float4 v = src[j];
        dst[j * 4 + 0] = (bf16)v.x; dst[j * 4 + 1] = (bf16)v.y;
        dst[j * 4 + 2] = (bf16)v.z; dst[j * 4 + 3] = (bf16)v.w;
      }
    }
    // B tile transposed: sB[n][k] = W[kt*32+k][col0+n] (float4 loads)
    {
      const int n4 = (t & 15) * 4;
      const int kb = t >> 4;     // 0..15
#pragma unroll
      for (int p = 0; p < 2; ++p) {
        const int k = kb + p * 16;
        const float4 v =
            *(const float4*)(W + (size_t)(kt * BK + k) * F_ + col0 + n4);
        sB[(n4 + 0) * LDSTR + k] = (bf16)v.x;
        sB[(n4 + 1) * LDSTR + k] = (bf16)v.y;
        sB[(n4 + 2) * LDSTR + k] = (bf16)v.z;
        sB[(n4 + 3) * LDSTR + k] = (bf16)v.w;
      }
    }
    __syncthreads();

    const bf16* aBase = sA + (wm * 32) * LDSTR;
    const bf16* bBase = sB + (wn * 32) * LDSTR;
    v16bf a0 = frag_a(aBase), a1 = frag_a(aBase + 16 * LDSTR);
    v16bf b0 = frag_b(bBase), b1 = frag_b(bBase + 16 * LDSTR);
    acc[0][0] = wmma_bf16f32(a0, b0, acc[0][0]);
    acc[0][1] = wmma_bf16f32(a0, b1, acc[0][1]);
    acc[1][0] = wmma_bf16f32(a1, b0, acc[1][0]);
    acc[1][1] = wmma_bf16f32(a1, b1, acc[1][1]);
  }

  const int cr = lane & 15, rh = lane >> 4;
#pragma unroll
  for (int mt = 0; mt < 2; ++mt)
#pragma unroll
    for (int nt = 0; nt < 2; ++nt)
#pragma unroll
      for (int r = 0; r < 8; ++r) {
        const int m = row0 + wm * 32 + mt * 16 + rh * 8 + r;
        const int n = col0 + wn * 32 + nt * 16 + cr;
        float v = acc[mt][nt][r];
        if (act) v = fmaxf(v, 0.0f) + 1.0f;
        out[(size_t)m * F_ + n] = (bf16)v;
      }
}

// ---- Kernel 2: denominator = sum_s K[b,s,f] --------------------------------
__global__ __launch_bounds__(256)
void ksum_kernel(const bf16* __restrict__ Kb, float* __restrict__ denom)
{
  const int f = blockIdx.x * 256 + threadIdx.x;
  const int b = blockIdx.y;
  const bf16* p = Kb + (size_t)b * S_ * F_ + f;
  float s = 0.0f;
  for (int i = 0; i < S_; ++i) s += (float)p[(size_t)i * F_];
  denom[b * F_ + f] = s;
}

// ---- Kernel 3: context[b,f,g] = (sum_s K[b,s,f]*V[b,s,g]) / (den[b,g]+eps) --
__global__ __launch_bounds__(NTHREADS)
void kv_context_kernel(const bf16* __restrict__ Kb, const bf16* __restrict__ Vb,
                       const float* __restrict__ denom, bf16* __restrict__ ctx)
{
  __shared__ bf16 sA[BM * LDSTR];
  __shared__ bf16 sB[BN * LDSTR];

  const int t = threadIdx.x;
  const int b = blockIdx.z;
  const size_t bo = (size_t)b * S_ * F_;
  const int f0 = blockIdx.y * BM;
  const int g0 = blockIdx.x * BN;
  const int w = t >> 5, wm = w & 3, wn = w >> 2, lane = t & 31;
  v8f acc[2][2] = {};

  for (int st = 0; st < S_ / BK; ++st) {
    if (st) __syncthreads();
    // A (transposed): sA[f][s] = K[b][st*32+s][f0+f]; b128 loads along f
    {
      const int f8 = (t & 15) * 8;     // 0..120
      const int sb = t >> 4;           // 0..15
#pragma unroll
      for (int p = 0; p < 2; ++p) {
        const int s = sb + p * 16;
        v8bf v = *(const v8bf*)(Kb + bo + (size_t)(st * BK + s) * F_ + f0 + f8);
#pragma unroll
        for (int j = 0; j < 8; ++j) sA[(f8 + j) * LDSTR + s] = v[j];
      }
    }
    // B (transposed): sB[g][s] = V[b][st*32+s][g0+g]; b128 loads along g
    {
      const int g8 = (t & 7) * 8;      // 0..56
      const int s  = t >> 3;           // 0..31
      v8bf v = *(const v8bf*)(Vb + bo + (size_t)(st * BK + s) * F_ + g0 + g8);
#pragma unroll
      for (int j = 0; j < 8; ++j) sB[(g8 + j) * LDSTR + s] = v[j];
    }
    __syncthreads();

    const bf16* aBase = sA + (wm * 32) * LDSTR;
    const bf16* bBase = sB + (wn * 32) * LDSTR;
    v16bf a0 = frag_a(aBase), a1 = frag_a(aBase + 16 * LDSTR);
    v16bf b0 = frag_b(bBase), b1 = frag_b(bBase + 16 * LDSTR);
    acc[0][0] = wmma_bf16f32(a0, b0, acc[0][0]);
    acc[0][1] = wmma_bf16f32(a0, b1, acc[0][1]);
    acc[1][0] = wmma_bf16f32(a1, b0, acc[1][0]);
    acc[1][1] = wmma_bf16f32(a1, b1, acc[1][1]);
  }

  const int cr = lane & 15, rh = lane >> 4;
#pragma unroll
  for (int nt = 0; nt < 2; ++nt) {
    const int g = g0 + wn * 32 + nt * 16 + cr;
    const float inv = 1.0f / (denom[b * F_ + g] + EPS_);
#pragma unroll
    for (int mt = 0; mt < 2; ++mt)
#pragma unroll
      for (int r = 0; r < 8; ++r) {
        const int f = f0 + wm * 32 + mt * 16 + rh * 8 + r;
        ctx[(size_t)b * F_ * F_ + (size_t)f * F_ + g] =
            (bf16)(acc[mt][nt][r] * inv);
      }
  }
}

// ---- Kernel 4: out[b,f,s] = sum_g ctx[b,f,g] * Q[b,s,g] --------------------
// Both LDS tiles are pure row-major copies -> Tensor Data Mover, double-buffered.
__global__ __launch_bounds__(NTHREADS)
void out_kernel(const bf16* __restrict__ ctx, const bf16* __restrict__ Qb,
                float* __restrict__ out)
{
  __shared__ bf16 sA[2][BM * LDSTR];
  __shared__ bf16 sB[2][BN * LDSTR];

  const int t = threadIdx.x;
  const int b = blockIdx.z;
  const int f0 = blockIdx.y * BM;
  const int s0 = blockIdx.x * BN;
  const size_t cbo = (size_t)b * F_ * F_;
  const size_t qbo = (size_t)b * S_ * F_;
  const int w = t >> 5, wm = w & 3, wn = w >> 2, lane = t & 31;
  const int NT = F_ / BK;   // 32 K-tiles
  v8f acc[2][2] = {};

  // TDM issue for tile gt into buffer buf (one instruction per tile per matrix)
  auto issue = [&](int gt, int buf) {
    tdm_load_tile_2d(lds_off(&sA[buf][0]),
                     ctx + cbo + (size_t)f0 * F_ + gt * BK,
                     F_, F_, BK, BM, F_);
    tdm_load_tile_2d(lds_off(&sB[buf][0]),
                     Qb + qbo + (size_t)s0 * F_ + gt * BK,
                     F_, S_, BK, BN, F_);
  };

  if (t < 32) {   // wave 0 drives the TDM (TDM ignores EXEC; branch per-wave)
    issue(0, 0);
    issue(1, 1);
  }

  for (int gt = 0; gt < NT; ++gt) {
    if (t < 32) {
      if (gt == NT - 1) __builtin_amdgcn_s_wait_tensorcnt(0);
      else              __builtin_amdgcn_s_wait_tensorcnt(2);
    }
    __syncthreads();   // tile gt resident in LDS for all waves

    const bf16* aBase = &sA[gt & 1][0] + (wm * 32) * LDSTR;
    const bf16* bBase = &sB[gt & 1][0] + (wn * 32) * LDSTR;
    v16bf a0 = frag_a(aBase), a1 = frag_a(aBase + 16 * LDSTR);
    v16bf b0 = frag_b(bBase), b1 = frag_b(bBase + 16 * LDSTR);
    acc[0][0] = wmma_bf16f32(a0, b0, acc[0][0]);
    acc[0][1] = wmma_bf16f32(a0, b1, acc[0][1]);
    acc[1][0] = wmma_bf16f32(a1, b0, acc[1][0]);
    acc[1][1] = wmma_bf16f32(a1, b1, acc[1][1]);

    __syncthreads();   // all waves done reading buf (gt&1) before refill
    if (t < 32 && gt + 2 < NT) issue(gt + 2, gt & 1);
  }

  const int cr = lane & 15, rh = lane >> 4;
#pragma unroll
  for (int mt = 0; mt < 2; ++mt)
#pragma unroll
    for (int nt = 0; nt < 2; ++nt)
#pragma unroll
      for (int r = 0; r < 8; ++r) {
        const int f = f0 + wm * 32 + mt * 16 + rh * 8 + r;
        const int s = s0 + wn * 32 + nt * 16 + cr;
        out[(size_t)b * F_ * S_ + (size_t)f * S_ + s] = acc[mt][nt][r];
      }
}

// ---- Host launcher ----------------------------------------------------------
extern "C" void kernel_launch(void* const* d_in, const int* in_sizes, int n_in,
                              void* d_out, int out_size, void* d_ws,
                              size_t ws_size, hipStream_t stream)
{
  const float* x  = (const float*)d_in[0];
  const float* Wq = (const float*)d_in[1];
  const float* Wk = (const float*)d_in[2];
  const float* Wv = (const float*)d_in[3];
  float* out = (float*)d_out;

  // workspace: Q,K,V bf16 (64MB each) | denom f32 (32KB) | ctx bf16 (16MB)
  const size_t nQKV = (size_t)B_ * S_ * F_;
  bf16* Qb = (bf16*)d_ws;
  bf16* Kb = Qb + nQKV;
  bf16* Vb = Kb + nQKV;
  float* denom = (float*)(Vb + nQKV);
  bf16* ctx = (bf16*)(denom + B_ * F_);

  dim3 blk(NTHREADS);
  qkv_proj_kernel<<<dim3(F_ / BN, (B_ * S_) / BM, 3), blk, 0, stream>>>(
      x, Wq, Wk, Wv, Qb, Kb, Vb);
  ksum_kernel<<<dim3(F_ / 256, B_), 256, 0, stream>>>(Kb, denom);
  kv_context_kernel<<<dim3(F_ / BN, F_ / BM, B_), blk, 0, stream>>>(
      Kb, Vb, denom, ctx);
  out_kernel<<<dim3(S_ / BN, F_ / BM, B_), blk, 0, stream>>>(ctx, Qb, out);
}
